// Our_tuning_for_transformer_13657996002002
// MI455X (gfx1250) — compile-verified, matrix operand order checked by reading, not run on hardware
//
#include <hip/hip_runtime.h>
#include <stdint.h>

// ---------------- constants (reference shapes) ----------------
constexpr int L_ = 13, B_ = 32, N_ = 512, H_ = 768, M_ = 384, E_ = 768;

// ---------------- WMMA types / helpers ----------------
typedef __attribute__((ext_vector_type(16))) __bf16          v16bf;
typedef __attribute__((ext_vector_type(8)))  float           v8f;
typedef __attribute__((ext_vector_type(8)))  unsigned short  u16x8;

union FragBF { v16bf v; u16x8 h[2]; unsigned short u[16]; };

__device__ __forceinline__ unsigned short f2bf(float f) {
  union { float f; unsigned u; } c; c.f = f;
  unsigned r = c.u + 0x7FFFu + ((c.u >> 16) & 1u);   // round-to-nearest-even
  return (unsigned short)(r >> 16);
}
__device__ __forceinline__ float bf2f(unsigned short h) {
  union { unsigned u; float f; } c; c.u = ((unsigned)h) << 16;
  return c.f;
}

// Async global->LDS 16B copy (CDNA5 GLOBAL_LOAD_ASYNC_TO_LDS_B128, ASYNCcnt).
// Generic pointers to LDS carry the LDS byte offset in their low 32 bits
// (ISA 10.2 aperture resolution), which is exactly the VDST operand.
__device__ __forceinline__ void async_ld16(void* lds, const void* g) {
  unsigned l = (unsigned)(size_t)lds;
  unsigned long long ga = (unsigned long long)(size_t)g;
  asm volatile("global_load_async_to_lds_b128 %0, %1, off"
               :: "v"(l), "v"(ga) : "memory");
}
__device__ __forceinline__ void async_wait() {
  asm volatile("s_wait_asynccnt 0x0" ::: "memory");
}

// Fragment (A or transposed-B) load: per ISA 7.12.2 the 16-bit fragment's K
// indices are contiguous per lane-half: e<8 -> k = lh*8+e ; e>=8 -> 16+lh*8+(e-8)
// => two 16-byte LDS loads (ds_load_b128). `base` must be 16B aligned.
__device__ __forceinline__ v16bf ld_frag(const unsigned short* base, int lh) {
  FragBF f;
  f.h[0] = *(const u16x8*)(base + lh * 8);
  f.h[1] = *(const u16x8*)(base + 16 + lh * 8);
  return f.v;
}

#define WMMA_BF16(A, Bm, C) \
  __builtin_amdgcn_wmma_f32_16x16x32_bf16(false, (A), false, (Bm), (short)0, (C), false, false)

// B-tile staging layout: transposed, buf[col*40 + kk], kk = 0..31 (K dim).
// Tile t (16 cols) fragment base = &buf[(t*16 + ll)*40].
constexpr int BSTR = 40;    // u16 stride: mult of 8 (16B align), 80B rotates banks

// =============================================================
// Kernel A: down = LN( X @ W_down + b_down ) per layer, bf16 out
// block: 16 rows x 384 cols, 256 thr (8 waves x 3 col-tiles)
// =============================================================
__global__ __launch_bounds__(256)
void kA_down_ln(const float* __restrict__ X, const float* __restrict__ Wd,
                const float* __restrict__ bd, const float* __restrict__ gd,
                const float* __restrict__ betad, unsigned short* __restrict__ down)
{
  __shared__ __align__(16) unsigned short sX[16 * BSTR];        // A rows (K-contig)
  __shared__ __align__(16) unsigned short sBW[384 * BSTR];      // B transposed
  __shared__ float redS[16], redQ[16];

  const int tid = threadIdx.x, wave = tid >> 5, lane = tid & 31;
  const int ll = lane & 15, lh = lane >> 4;

  const int bid = blockIdx.x;
  const int l   = bid / (B_ * 32);
  const int rem = bid % (B_ * 32);
  const int b   = rem >> 5;
  const int n0  = (rem & 31) * 16;

  const size_t xbase = (((size_t)l * B_ + b) * N_ + n0) * H_;
  const size_t wbase = (size_t)l * H_ * M_;

  const int kkp = tid >> 4, kk2 = kkp * 2, mlo = tid & 15;

  const v8f zero = {0.f,0.f,0.f,0.f,0.f,0.f,0.f,0.f};
  v8f acc[3] = {zero, zero, zero};

  for (int ks = 0; ks < H_ / 32; ++ks) {
    const int k0 = ks * 32;
    __syncthreads();
    // X tile 16x32 : float2 global loads -> packed u32 LDS stores
    {
      int r = tid >> 4, c = (tid & 15) * 2;   // 256 pairs, one per thread
      float2 xv = *(const float2*)&X[xbase + (size_t)r * H_ + k0 + c];
      unsigned pk = (unsigned)f2bf(xv.x) | ((unsigned)f2bf(xv.y) << 16);
      *(unsigned*)&sX[r * BSTR + c] = pk;
    }
    // W tile 32x384 transposed: pair along kk (kk fixed per thread, no division)
    for (int m = mlo; m < 384; m += 16) {
      unsigned pk = (unsigned)f2bf(Wd[wbase + (size_t)(k0 + kk2) * M_ + m])
                  | ((unsigned)f2bf(Wd[wbase + (size_t)(k0 + kk2 + 1) * M_ + m]) << 16);
      *(unsigned*)&sBW[m * BSTR + kk2] = pk;
    }
    if (ks + 1 < H_ / 32) __builtin_prefetch(&X[xbase + k0 + 32], 0, 0);
    __syncthreads();

    v16bf a = ld_frag(&sX[ll * BSTR], lh);
#pragma unroll
    for (int t = 0; t < 3; ++t) {
      v16bf bm = ld_frag(&sBW[(wave * 48 + t * 16 + ll) * BSTR], lh);
      acc[t] = WMMA_BF16(a, bm, acc[t]);
    }
  }

  // bias + LayerNorm over the 384 features of each of the 16 rows
  if (tid < 16) { redS[tid] = 0.f; redQ[tid] = 0.f; }
  __syncthreads();
  float xv[3][8];
#pragma unroll
  for (int t = 0; t < 3; ++t) {
    const int m0 = wave * 48 + t * 16 + ll;
    const float bb = bd[l * M_ + m0];
#pragma unroll
    for (int r = 0; r < 8; ++r) {
      float v = acc[t][r] + bb;
      xv[t][r] = v;
      const int nl = r + 8 * lh;
      atomicAdd(&redS[nl], v);
      atomicAdd(&redQ[nl], v * v);
    }
  }
  __syncthreads();
#pragma unroll
  for (int t = 0; t < 3; ++t) {
    const int m0 = wave * 48 + t * 16 + ll;
    const float gv = gd[l * M_ + m0], bv = betad[l * M_ + m0];
#pragma unroll
    for (int r = 0; r < 8; ++r) {
      const int nl = r + 8 * lh;
      const float mean = redS[nl] * (1.f / 384.f);
      const float var  = redQ[nl] * (1.f / 384.f) - mean * mean;
      const float y = (xv[t][r] - mean) * rsqrtf(var + 1e-5f) * gv + bv;
      down[(((size_t)l * B_ + b) * N_ + n0 + nl) * M_ + m0] = f2bf(y);
    }
  }
}

// =============================================================
// Kernel B: agg[l,b,m] = masked mean over n of down[l,b,n,m]
// =============================================================
__global__ __launch_bounds__(384)
void kB_agg(const unsigned short* __restrict__ down,
            const int* __restrict__ lengths, float* __restrict__ agg)
{
  const int l = blockIdx.x >> 5;
  const int b = blockIdx.x & 31;
  const int m = threadIdx.x;
  const int len = lengths[b];
  const unsigned short* base = down + (((size_t)l * B_ + b) * N_) * M_ + m;
  float s = 0.f;
  for (int n = 0; n < len; ++n) s += bf2f(base[(size_t)n * M_]);
  agg[((size_t)l * B_ + b) * M_ + m] = s / (float)len;
}

// =============================================================
// Kernel C: logits + softmax over l -> conf[b*12+l]
// =============================================================
__global__ __launch_bounds__(384)
void kC_conf(const float* __restrict__ agg, const float* __restrict__ Wg,
             const float* __restrict__ bg, float* __restrict__ conf)
{
  __shared__ float lg[384];
  const int t = threadIdx.x;
  const int l = t / 32, b = t & 31;
  const float* anchor = agg + ((size_t)12 * B_ + b) * M_;
  const float* ta     = agg + ((size_t)l  * B_ + b) * M_;
  float s = 0.f;
  for (int m = 0; m < M_; ++m) s += anchor[m] * ta[m] * Wg[m];
  lg[l * 32 + b] = s + bg[0];
  __syncthreads();
  if (t < 32) {
    const int bb = t;
    float mx = -1e30f;
    for (int l2 = 0; l2 < 12; ++l2) mx = fmaxf(mx, lg[l2 * 32 + bb]);
    float ev[12], sum = 0.f;
    for (int l2 = 0; l2 < 12; ++l2) { ev[l2] = __expf(lg[l2 * 32 + bb] - mx); sum += ev[l2]; }
    for (int l2 = 0; l2 < 12; ++l2) conf[bb * 12 + l2] = ev[l2] / sum;
  }
}

// =============================================================
// Kernel D: fused scores -> relu/mask/normalize/+I -> merged -> msum
// block: (b, 16-row tile). loops l=0..11, accumulates conf-weighted merged.
// =============================================================
__global__ __launch_bounds__(256)
void kD_merge(const unsigned short* __restrict__ down, const int* __restrict__ lengths,
              const float* __restrict__ conf, unsigned short* __restrict__ msum)
{
  __shared__ __align__(16) unsigned short sA[16 * 392];       // anchor rows (bf16)
  __shared__ __align__(16) unsigned short sStage[512 * BSTR]; // B tiles (transposed)
  __shared__ float          sS[16 * 520];                     // score rows f32
  __shared__ __align__(16) unsigned short scw[16 * 520];      // normalized cw bf16
  __shared__ float          sInv[16];

  const int tid = threadIdx.x, wave = tid >> 5, lane = tid & 31;
  const int ll = lane & 15, lh = lane >> 4;
  const int b  = blockIdx.x >> 5;
  const int n0 = (blockIdx.x & 31) * 16;
  const int len = lengths[b];
  const int kkp = tid >> 4, kk2 = kkp * 2, mlo = tid & 15;

  // anchor rows: pure bf16 copy -> async DMA to LDS (16 rows x 48 x 16B chunks)
  const size_t anchorBase = (((size_t)12 * B_ + b) * N_ + n0) * M_;
  {
    const int r = tid & 15;
#pragma unroll
    for (int j = 0; j < 3; ++j) {
      const int chunk = (tid >> 4) + j * 16;      // 0..47
      async_ld16(&sA[r * 392 + chunk * 8],
                 &down[anchorBase + (size_t)r * M_ + chunk * 8]);
    }
  }

  const v8f zero = {0.f,0.f,0.f,0.f,0.f,0.f,0.f,0.f};
  v8f accM[3] = {zero, zero, zero};

  for (int l = 0; l < 12; ++l) {
    const size_t tBase = (((size_t)l * B_ + b) * N_) * M_;

    // ---- scores S[16 x 512] = anchor_rows @ T^T (K over m) ----
    v8f accS[4] = {zero, zero, zero, zero};
    for (int ms = 0; ms < 12; ++ms) {
      const int m0 = ms * 32;
      __syncthreads();
      // stage T[r=0..511, m0..m0+31] as rows of 64B: async DMA, no VGPR round-trip
#pragma unroll
      for (int q = tid; q < 512 * 4; q += 256) {
        const int r = q >> 2, part = q & 3;
        async_ld16(&sStage[r * BSTR + part * 8],
                   &down[tBase + (size_t)r * M_ + m0 + part * 8]);
      }
      async_wait();
      __syncthreads();
      v16bf a = ld_frag(&sA[ll * 392 + m0], lh);
#pragma unroll
      for (int j = 0; j < 4; ++j) {
        const int kt = wave * 4 + j;
        v16bf bm = ld_frag(&sStage[(kt * 16 + ll) * BSTR], lh);
        accS[j] = WMMA_BF16(a, bm, accS[j]);
      }
    }
    __syncthreads();
#pragma unroll
    for (int j = 0; j < 4; ++j) {
      const int kc = (wave * 4 + j) * 16 + ll;
#pragma unroll
      for (int r = 0; r < 8; ++r) {
        const int nl = r + 8 * lh;
        float v = accS[j][r];
        v = (kc < len) ? fmaxf(v, 0.f) : 0.f;   // relu(cw * mask)
        sS[nl * 520 + kc] = v;
      }
    }
    __syncthreads();
    if (tid < 16) {
      float s = 0.f;
      for (int k = 0; k < 512; ++k) s += sS[tid * 520 + k];
      sInv[tid] = 1.f / (s + 1e-8f);
    }
    __syncthreads();
    for (int i = tid; i < 16 * 512; i += 256) {
      int n = i >> 9, k = i & 511;
      float v = sS[n * 520 + k] * sInv[n] + (((n0 + n) == k) ? 1.f : 0.f);
      scw[n * 520 + k] = f2bf(v);
    }
    const float cf = conf[b * 12 + l];

    // ---- merged[16 x 384] = cw @ T (K over k=512) ----
    v8f accT[3] = {zero, zero, zero};
    for (int ksn = 0; ksn < 16; ++ksn) {
      const int k0 = ksn * 32;
      __syncthreads();
      // stage T[k0+kk, m] transposed-as-B: pair along kk (fixed per thread)
      for (int m = mlo; m < 384; m += 16) {
        unsigned lo = down[tBase + (size_t)(k0 + kk2) * M_ + m];
        unsigned hi = down[tBase + (size_t)(k0 + kk2 + 1) * M_ + m];
        *(unsigned*)&sStage[m * BSTR + kk2] = lo | (hi << 16);
      }
      __syncthreads();
      v16bf a = ld_frag(&scw[ll * 520 + k0], lh);
#pragma unroll
      for (int t = 0; t < 3; ++t) {
        v16bf bm = ld_frag(&sStage[(wave * 48 + t * 16 + ll) * BSTR], lh);
        accT[t] = WMMA_BF16(a, bm, accT[t]);
      }
    }
#pragma unroll
    for (int t = 0; t < 3; ++t)
#pragma unroll
      for (int r = 0; r < 8; ++r) accM[t][r] += cf * accT[t][r];
  }

#pragma unroll
  for (int t = 0; t < 3; ++t) {
    const int m0 = wave * 48 + t * 16 + ll;
#pragma unroll
    for (int r = 0; r < 8; ++r) {
      const int nl = r + 8 * lh;
      msum[((size_t)b * N_ + n0 + nl) * M_ + m0] = f2bf(accM[t][r]);
    }
  }
}

// =============================================================
// Kernel E: out = relu(anchor + LN(msum @ Wp + bp)) @ Wu + bu
// =============================================================
__global__ __launch_bounds__(256)
void kE_out(const unsigned short* __restrict__ msum, const unsigned short* __restrict__ down,
            const float* __restrict__ Wp, const float* __restrict__ bp,
            const float* __restrict__ gp, const float* __restrict__ betap,
            const float* __restrict__ Wu, const float* __restrict__ bu,
            const int* __restrict__ lengths, float* __restrict__ out)
{
  __shared__ __align__(16) unsigned short sStage[768 * BSTR]; // B tiles (transposed)
  __shared__ __align__(16) unsigned short sIn[16 * BSTR];
  __shared__ __align__(16) unsigned short sR[16 * 392];
  __shared__ float redS[16], redQ[16];

  const int tid = threadIdx.x, wave = tid >> 5, lane = tid & 31;
  const int ll = lane & 15, lh = lane >> 4;
  const int row0 = blockIdx.x * 16;
  const int b = row0 / N_, n0 = row0 - b * N_;
  const int kkp = tid >> 4, kk2 = kkp * 2, mlo = tid & 15;

  const v8f zero = {0.f,0.f,0.f,0.f,0.f,0.f,0.f,0.f};
  v8f acc[3] = {zero, zero, zero};
  const size_t inBase = (size_t)row0 * M_;

  for (int ks = 0; ks < 12; ++ks) {
    const int k0 = ks * 32;
    __syncthreads();
    // msum A tile: pure bf16 copy -> async DMA (16 rows x 4 x 16B)
    if (tid < 64) {
      const int r = tid >> 2, part = tid & 3;
      async_ld16(&sIn[r * BSTR + part * 8],
                 &msum[inBase + (size_t)r * M_ + k0 + part * 8]);
    }
    // Wp transposed: pair along kk (fixed per thread)
    for (int m = mlo; m < 384; m += 16) {
      unsigned pk = (unsigned)f2bf(Wp[(size_t)(k0 + kk2) * M_ + m])
                  | ((unsigned)f2bf(Wp[(size_t)(k0 + kk2 + 1) * M_ + m]) << 16);
      *(unsigned*)&sStage[m * BSTR + kk2] = pk;
    }
    async_wait();
    __syncthreads();
    v16bf a = ld_frag(&sIn[ll * BSTR], lh);
#pragma unroll
    for (int t = 0; t < 3; ++t) {
      v16bf bm = ld_frag(&sStage[(wave * 48 + t * 16 + ll) * BSTR], lh);
      acc[t] = WMMA_BF16(a, bm, acc[t]);
    }
  }

  // bias + LN + anchor + relu -> sR (bf16)
  if (tid < 16) { redS[tid] = 0.f; redQ[tid] = 0.f; }
  __syncthreads();
  float xv[3][8];
#pragma unroll
  for (int t = 0; t < 3; ++t) {
    const int m0 = wave * 48 + t * 16 + ll;
    const float bb = bp[m0];
#pragma unroll
    for (int r = 0; r < 8; ++r) {
      float v = acc[t][r] + bb;
      xv[t][r] = v;
      const int nl = r + 8 * lh;
      atomicAdd(&redS[nl], v);
      atomicAdd(&redQ[nl], v * v);
    }
  }
  __syncthreads();
#pragma unroll
  for (int t = 0; t < 3; ++t) {
    const int m0 = wave * 48 + t * 16 + ll;
    const float gv = gp[m0], bv = betap[m0];
#pragma unroll
    for (int r = 0; r < 8; ++r) {
      const int nl = r + 8 * lh;
      const float mean = redS[nl] * (1.f / 384.f);
      const float var  = redQ[nl] * (1.f / 384.f) - mean * mean;
      const float y = (xv[t][r] - mean) * rsqrtf(var + 1e-5f) * gv + bv;
      const float anc = bf2f(down[(((size_t)12 * B_ + b) * N_ + n0 + nl) * M_ + m0]);
      sR[nl * 392 + m0] = f2bf(fmaxf(anc + y, 0.f));
    }
  }
  __syncthreads();

  // up-projection: 16 x 768 = sR(16x384) @ Wu(384x768)
  v8f accU[6] = {zero, zero, zero, zero, zero, zero};
  for (int ks = 0; ks < 12; ++ks) {
    const int k0 = ks * 32;
    __syncthreads();
    for (int m = mlo; m < 768; m += 16) {          // Wu transposed, pair along kk
      unsigned pk = (unsigned)f2bf(Wu[(size_t)(k0 + kk2) * E_ + m])
                  | ((unsigned)f2bf(Wu[(size_t)(k0 + kk2 + 1) * E_ + m]) << 16);
      *(unsigned*)&sStage[m * BSTR + kk2] = pk;
    }
    __syncthreads();
    v16bf a = ld_frag(&sR[ll * 392 + k0], lh);
#pragma unroll
    for (int t = 0; t < 6; ++t) {
      v16bf bm = ld_frag(&sStage[(wave * 96 + t * 16 + ll) * BSTR], lh);
      accU[t] = WMMA_BF16(a, bm, accU[t]);
    }
  }
#pragma unroll
  for (int t = 0; t < 6; ++t) {
    const int m0 = wave * 96 + t * 16 + ll;
    const float bb = bu[m0];
#pragma unroll
    for (int r = 0; r < 8; ++r) {
      const int nl = r + 8 * lh;
      out[(size_t)(row0 + nl) * E_ + m0] = accU[t][r] + bb;
    }
  }
  // append lengths (int bits) after the f32 tensor
  if (blockIdx.x == 0 && tid < 32) {
    int* oi = (int*)out;
    oi[(size_t)B_ * N_ * E_ + tid] = lengths[tid];
  }
}

// =============================================================
extern "C" void kernel_launch(void* const* d_in, const int* in_sizes, int n_in,
                              void* d_out, int out_size, void* d_ws, size_t ws_size,
                              hipStream_t stream)
{
  (void)in_sizes; (void)n_in; (void)out_size; (void)ws_size;
  const float* X      = (const float*)d_in[0];
  const int*   lens   = (const int*)  d_in[1];
  const float* Wd     = (const float*)d_in[2];
  const float* bd     = (const float*)d_in[3];
  const float* gd     = (const float*)d_in[4];
  const float* betad  = (const float*)d_in[5];
  const float* Wp     = (const float*)d_in[6];
  const float* bp     = (const float*)d_in[7];
  const float* gp     = (const float*)d_in[8];
  const float* betap  = (const float*)d_in[9];
  const float* Wu     = (const float*)d_in[10];
  const float* bu     = (const float*)d_in[11];
  const float* Wg     = (const float*)d_in[12];
  const float* bg     = (const float*)d_in[13];

  char* ws = (char*)d_ws;
  const size_t downBytes = (size_t)L_ * B_ * N_ * M_ * 2;        // 163.6 MB bf16
  const size_t aggBytes  = (size_t)L_ * B_ * M_ * 4;             // 0.64 MB f32
  unsigned short* down = (unsigned short*)ws;
  float*          agg  = (float*)(ws + downBytes);
  float*          conf = (float*)(ws + downBytes + aggBytes);
  unsigned short* msum = (unsigned short*)(ws + downBytes + aggBytes + 2048);
  float* out = (float*)d_out;

  kA_down_ln<<<dim3(L_ * B_ * (N_ / 16)), dim3(256), 0, stream>>>(X, Wd, bd, gd, betad, down);
  kB_agg    <<<dim3(L_ * B_),             dim3(384), 0, stream>>>(down, lens, agg);
  kC_conf   <<<dim3(1),                   dim3(384), 0, stream>>>(agg, Wg, bg, conf);
  kD_merge  <<<dim3(B_ * (N_ / 16)),      dim3(256), 0, stream>>>(down, lens, conf, msum);
  kE_out    <<<dim3((B_ * N_) / 16),      dim3(256), 0, stream>>>(msum, down, Wp, bp, gp, betap,
                                                                  Wu, bu, lens, out);
}
